// HybridDual_27771258536773
// MI455X (gfx1250) — matrix operand order repaired and checked by reading, not compiled, not run on hardware
//
#include <hip/hip_runtime.h>
#include <stdint.h>

typedef __attribute__((ext_vector_type(16))) _Float16 v16h;
typedef __attribute__((ext_vector_type(8)))  float    v8f;
typedef __attribute__((ext_vector_type(4)))  int      v4i;
typedef _Float16 half_t;

#if defined(__has_builtin)
#if __has_builtin(__builtin_amdgcn_global_load_async_to_lds_b128) && \
    __has_builtin(__builtin_amdgcn_s_wait_asynccnt)
#define HAVE_ASYNC_LDS 1
#endif
#endif
#ifndef HAVE_ASYNC_LDS
#define HAVE_ASYNC_LDS 0
#endif

typedef __attribute__((address_space(1))) v4i as1_v4i;   // global int4
typedef __attribute__((address_space(3))) v4i as3_v4i;   // LDS int4

// ---------------------------------------------------------------------------
// Pack OIHW f32 weights into per-lane WMMA A-fragment layout (f16, dwords).
// One block per Cout/16 chunk. Layout: [tap*NCHUNK+chunk][lane][8 dwords].
// ---------------------------------------------------------------------------
template <int KSZ, int CINBUF>
__global__ __launch_bounds__(256)
void pack_weights_kernel(const float* __restrict__ wgt, uint32_t* __restrict__ wpack,
                         int CinW, int Cout)
{
  constexpr int NCHUNK = (CINBUF + 31) / 32;
  constexpr int NTAP   = KSZ * KSZ;
  constexpr int TOT    = NTAP * NCHUNK * 256;
  const int coBase = blockIdx.x << 4;
  uint32_t* dst = wpack + (size_t)blockIdx.x * TOT;
  for (int idx = threadIdx.x; idx < TOT; idx += 256) {
    const int d     = idx & 7;                 // dword within lane's 8
    const int ln    = (idx >> 3) & 31;         // lane
    const int tc    = idx >> 8;                // tap*NCHUNK + chunk
    const int tap   = tc / NCHUNK;             // NCHUNK is 1 or 2 -> shift
    const int chunk = tc - tap * NCHUNK;
    const int kh    = tap / KSZ, kw = tap - kh * KSZ;
    const int mm    = ln & 15, lh = ln >> 4;
    const int co    = coBase + mm;
    uint32_t pk = 0;
#pragma unroll
    for (int s = 0; s < 2; ++s) {
      const int h  = 2 * d + s;
      const int K  = h + (lh ? 16 : 0);  // contiguous k = 16*half + s
      const int ci = (chunk << 5) + K;
      float wvv = 0.f;
      if ((co < Cout) && (ci < CinW))
        wvv = wgt[((co * CinW + ci) * KSZ + kh) * KSZ + kw];
      union { _Float16 hh; uint16_t us; } cv; cv.hh = (_Float16)wvv;
      pk |= ((uint32_t)cv.us) << (16 * s);
    }
    dst[idx] = pk;
  }
}

// ---------------------------------------------------------------------------
// Implicit-GEMM conv (3x3 pad1 / 1x1 pad0), fused bias/BN/skip/ReLU.
// Activations NHWC f16, weights pre-packed WMMA A-fragments (async-copied to
// LDS), f32 accumulation via v_wmma_f32_16x16x32_f16.
// Grid: x = pixel-tile octets, y = Cout/16 chunk, z = batch. 8 waves/block,
// one wave -> 16(Cout) x 16(pixels along W) tile. All index math shift/mask.
// ---------------------------------------------------------------------------
template <int KSZ, int CINBUF>
__global__ __launch_bounds__(256)
void conv_wmma_kernel(const half_t* __restrict__ in,
                      const uint32_t* __restrict__ wpack,
                      const float* __restrict__ bias,
                      const float* __restrict__ scale,
                      const float* __restrict__ shift,
                      const half_t* __restrict__ skip,
                      half_t* __restrict__ out,
                      int H, int W, int tileShift,   // tileShift = log2(W/16)
                      int Cout, int relu)
{
  constexpr int NCHUNK = (CINBUF + 31) / 32;
  constexpr int NTAP   = KSZ * KSZ;
  constexpr int PAD    = KSZ / 2;
  constexpr int TOTDW  = NTAP * NCHUNK * 256;
  __shared__ alignas(32) uint32_t ldsW[TOTDW];

  const int coBase = blockIdx.y << 4;
  const int b      = blockIdx.z;
  const int lane   = threadIdx.x & 31;
  const int wv     = threadIdx.x >> 5;

  // ---- bring this coChunk's A-fragments into LDS (async on gfx1250) ------
  const uint32_t* gsrc = wpack + (size_t)blockIdx.y * TOTDW;
#if HAVE_ASYNC_LDS
  for (int d4 = threadIdx.x * 4; d4 < TOTDW; d4 += 1024) {
    __builtin_amdgcn_global_load_async_to_lds_b128(
        (as1_v4i*)(gsrc + d4), (as3_v4i*)(ldsW + d4), 0, 0);
  }
  __builtin_amdgcn_s_wait_asynccnt(0);
#else
  for (int idx = threadIdx.x; idx < TOTDW; idx += 256) ldsW[idx] = gsrc[idx];
#endif
  __syncthreads();

  const int tilesPerImg = H << tileShift;
  const int tile = (int)(blockIdx.x << 3) + wv;
  if (tile >= tilesPerImg) return;              // wave-uniform, after barrier

  const int py     = tile >> tileShift;
  const int pxBase = (tile & ((1 << tileShift) - 1)) << 4;
  const int m      = lane & 15;
  const int hi     = lane >> 4;

  v8f acc = {};

#pragma unroll
  for (int tap = 0; tap < NTAP; ++tap) {
    const int kh = tap / KSZ, kw = tap - kh * KSZ;
    const int iy = py + kh - PAD;
    const int ix = pxBase + m + kw - PAD;
    const bool inb = (iy >= 0) && (iy < H) && (ix >= 0) && (ix < W);
    const size_t rowBase =
        ((size_t)(b * H + (inb ? iy : 0)) * W + (inb ? ix : 0)) * (size_t)CINBUF;
#pragma unroll
    for (int chunk = 0; chunk < NCHUNK; ++chunk) {
      const v16h afrag =
          *(const v16h*)(ldsW + (((size_t)(tap * NCHUNK + chunk) * 32 + lane) * 8));
      v16h bfrag = {};
      if constexpr (CINBUF >= 32) {
        if (inb) {
          union { uint4 q[2]; v16h v; } t;
          const uint4* p = (const uint4*)(in + rowBase + (chunk << 5) + (hi ? 16 : 0));
          t.q[0] = p[0]; t.q[1] = p[1];
          bfrag = t.v;
        }
      } else if constexpr (CINBUF == 16) {
        if (inb && !hi) {                        // lanes 0-15 carry K=0..15
          union { uint4 q[2]; v16h v; } t;
          const uint4* p = (const uint4*)(in + rowBase);
          t.q[0] = p[0]; t.q[1] = p[1];
          bfrag = t.v;
        }
      } else {                                   // CINBUF == 8
        if (inb && !hi) {
          union { uint4 q; _Float16 h[8]; } t;
          t.q = *(const uint4*)(in + rowBase);
#pragma unroll
          for (int h2 = 0; h2 < 8; ++h2) bfrag[h2] = t.h[h2];
        }
      }
      acc = __builtin_amdgcn_wmma_f32_16x16x32_f16(false, afrag, false, bfrag,
                                                   (short)0, acc, false, false);
    }
  }

  // gfx1250 prefetch of next pixel tile's data
  __builtin_prefetch(in + ((size_t)(b * H + py) * W + pxBase) * CINBUF
                        + (size_t)16 * CINBUF, 0, 1);

  // ---- fused epilogue: +bias -> BN -> +skip -> ReLU, one 16B store -------
  const int coOff = coBase + (hi ? 8 : 0);
  if (coOff + 8 <= Cout) {                       // Cout is a multiple of 8
    const size_t dstE = ((size_t)(b * H + py) * W + (pxBase + m)) * Cout + coOff;
    union { float4 v[2]; float a[8]; } bb, sc2, sh2;
    bb.v[0] = *(const float4*)(bias + coOff);
    bb.v[1] = *(const float4*)(bias + coOff + 4);
    if (scale) {
      sc2.v[0] = *(const float4*)(scale + coOff);
      sc2.v[1] = *(const float4*)(scale + coOff + 4);
      sh2.v[0] = *(const float4*)(shift + coOff);
      sh2.v[1] = *(const float4*)(shift + coOff + 4);
    }
    float sk[8];
    if (skip) {
      union { uint4 q; _Float16 h[8]; } sv;
      sv.q = *(const uint4*)(skip + dstE);
#pragma unroll
      for (int f = 0; f < 8; ++f) sk[f] = (float)sv.h[f];
    }
    union { _Float16 h[8]; uint4 q; } st;
#pragma unroll
    for (int f = 0; f < 8; ++f) {
      float r = acc[f] + bb.a[f];
      if (scale) r = r * sc2.a[f] + sh2.a[f];
      if (skip)  r += sk[f];
      if (relu)  r = fmaxf(r, 0.f);
      st.h[f] = (_Float16)r;
    }
    *(uint4*)(out + dstE) = st.q;
  }
}

// ---------------------------------------------------------------------------
__global__ void pack_input_kernel(const float* __restrict__ x,
                                  half_t* __restrict__ out, int total)
{
  int i = blockIdx.x * blockDim.x + threadIdx.x;
  if (i >= total) return;
  half_t* p = out + (size_t)i * 8;               // pad C=1 -> C=8 with zeros
  p[0] = (half_t)x[i];
#pragma unroll
  for (int c = 1; c < 8; ++c) p[c] = (half_t)0.f;
}

// shifts: cs=log2(C), wos=log2(Wo), hos=log2(Ho); all dims are powers of two
__global__ void maxpool_kernel(const half_t* __restrict__ in, half_t* __restrict__ out,
                               int H, int W, int C, int cs, int wos, int hos, int total)
{
  int i = blockIdx.x * blockDim.x + threadIdx.x;
  if (i >= total) return;
  const int Wo = W >> 1, Ho = H >> 1;
  int c = i & (C - 1); int t = i >> cs;
  int x = t & (Wo - 1); t >>= wos;
  int y = t & (Ho - 1); int b = t >> hos;
  const half_t* p = in + ((size_t)(b * H + 2 * y) * W + 2 * x) * C + c;
  float v0 = (float)p[0], v1 = (float)p[C];
  float v2 = (float)p[(size_t)W * C], v3 = (float)p[(size_t)W * C + C];
  out[i] = (half_t)fmaxf(fmaxf(v0, v1), fmaxf(v2, v3));
}

__global__ void upsample_kernel(const half_t* __restrict__ in, half_t* __restrict__ out,
                                int H, int W, int C, int cs, int wos, int hos, int total)
{
  int i = blockIdx.x * blockDim.x + threadIdx.x;
  if (i >= total) return;
  const int Wo = W << 1, Ho = H << 1;
  int c = i & (C - 1); int t = i >> cs;
  int x = t & (Wo - 1); t >>= wos;
  int y = t & (Ho - 1); int b = t >> hos;
  float sy = y * (float)(H - 1) / (float)(Ho - 1);
  float sx = x * (float)(W - 1) / (float)(Wo - 1);
  int y0 = (int)sy; int x0 = (int)sx;
  int y1 = (y0 + 1 < H) ? y0 + 1 : H - 1;
  int x1 = (x0 + 1 < W) ? x0 + 1 : W - 1;
  float wy = sy - y0, wx = sx - x0;
  const half_t* base = in + (size_t)b * H * W * C;
  float a00 = (float)base[((size_t)y0 * W + x0) * C + c];
  float a01 = (float)base[((size_t)y0 * W + x1) * C + c];
  float a10 = (float)base[((size_t)y1 * W + x0) * C + c];
  float a11 = (float)base[((size_t)y1 * W + x1) * C + c];
  out[i] = (half_t)((1.f - wy) * ((1.f - wx) * a00 + wx * a01)
                    + wy * ((1.f - wx) * a10 + wx * a11));
}

// 1x1 conv 8->4 at 512x512, f32 NCHW output (final head)
__global__ void conv_last_kernel(const half_t* __restrict__ in, const float* __restrict__ w,
                                 const float* __restrict__ bias, float* __restrict__ out)
{
  int i = blockIdx.x * blockDim.x + threadIdx.x;
  if (i >= 4 * 4 * 512 * 512) return;
  int x = i & 511; int y = (i >> 9) & 511; int c = (i >> 18) & 3; int b = i >> 20;
  const half_t* p = in + (((size_t)(b * 512 + y) << 9) + x) * 8;
  float r = bias[c];
#pragma unroll
  for (int ci = 0; ci < 8; ++ci) r += w[c * 8 + ci] * (float)p[ci];
  out[i] = r;
}

// mu[b,z] = h5_flat(NCHW) . W[z,:] + b[z]     (one block per output)
__global__ __launch_bounds__(256)
void fcmu_kernel(const half_t* __restrict__ h, const float* __restrict__ w,
                 const float* __restrict__ bias, float* __restrict__ mu)
{
  const int o = blockIdx.x;                      // b*64+z
  const int z = o & 63, b = o >> 6;
  const float* wr = w + (size_t)z * 65536;
  const half_t* hb = h + (size_t)b * 65536;
  float s = 0.f;
  for (int i = threadIdx.x; i < 65536; i += 256) {
    int c = i >> 10, r = i & 1023;
    int y = r >> 5, x = r & 31;
    s += wr[i] * (float)hb[(size_t)(((y << 5) + x) * 64 + c)];
  }
  __shared__ float red[256];
  red[threadIdx.x] = s;
  __syncthreads();
  for (int st = 128; st > 0; st >>= 1) {
    if (threadIdx.x < st) red[threadIdx.x] += red[threadIdx.x + st];
    __syncthreads();
  }
  if (threadIdx.x == 0) mu[o] = red[0] + bias[z];
}

__global__ void decfc_kernel(const float* __restrict__ mu, const float* __restrict__ w,
                             const float* __restrict__ bias, half_t* __restrict__ out)
{
  int i = blockIdx.x * blockDim.x + threadIdx.x;
  if (i >= 4 * 65536) return;
  int j = i & 65535, b = i >> 16;
  const float* m = mu + b * 64;
  const float* wr = w + (size_t)j * 64;
  float s = bias[j];
#pragma unroll 4
  for (int k = 0; k < 64; ++k) s += m[k] * wr[k];
  int c = j >> 10, r = j & 1023;
  int y = r >> 5, x = r & 31;
  out[((size_t)b * 1024 + (y << 5) + x) * 64 + c] = (half_t)s;   // NHWC f16
}

__global__ void declin_kernel(const float* __restrict__ mu, const float* __restrict__ w,
                              const float* __restrict__ bias, float* __restrict__ g)
{
  int i = blockIdx.x * blockDim.x + threadIdx.x;
  if (i >= 4 * 32768) return;
  int j = i & 32767, b = i >> 15;
  const float* m = mu + b * 64;
  const float* wr = w + (size_t)j * 64;
  float s = bias[j];
#pragma unroll 4
  for (int k = 0; k < 64; ++k) s += m[k] * wr[k];
  g[i] = fmaxf(s, 0.f);
}

__global__ void fill_kernel(float* __restrict__ p, float v, int n)
{ int i = blockIdx.x * blockDim.x + threadIdx.x; if (i < n) p[i] = v; }

// out[b,r,:] += val_e * x[b,c_e,:] ; grid.y = batch -> no division
__global__ void spmm_kernel(const int* __restrict__ rows, const int* __restrict__ cols,
                            const float* __restrict__ vals, const float* __restrict__ x,
                            float* __restrict__ out, int E, int F, int nIn, int nOut)
{
  int e = blockIdx.x * blockDim.x + threadIdx.x;
  if (e >= E) return;
  const int b = blockIdx.y;
  int r = rows[e], c = cols[e]; float v = vals[e];
  const float* xs = x + ((size_t)b * nIn + c) * F;
  float* os = out + ((size_t)b * nOut + r) * F;
  for (int f = 0; f < F; ++f) atomicAdd(os + f, v * xs[f]);
}

__global__ void cheb_combine_kernel(const float* __restrict__ S, const float* __restrict__ T0,
                                    float* __restrict__ T2, int n)
{ int i = blockIdx.x * blockDim.x + threadIdx.x; if (i < n) T2[i] = 2.f * S[i] - T0[i]; }

__global__ void chebmm_kernel(const float* __restrict__ T, const float* __restrict__ Wk,
                              float* __restrict__ out, int rows, int G, int gShift)
{
  int i = blockIdx.x * blockDim.x + threadIdx.x;
  if (i >= (rows << gShift)) return;
  int g = i & (G - 1); int r = i >> gShift;
  const float* t = T + (size_t)r * 16;
  float s = 0.f;
#pragma unroll 4
  for (int f = 0; f < 16; ++f) s += t[f] * Wk[f * G + g];
  out[i] += s;
}

__global__ void bias_relu_kernel(float* __restrict__ p, const float* __restrict__ bias, int n)
{ int i = blockIdx.x * blockDim.x + threadIdx.x; if (i < n) p[i] = fmaxf(p[i] + bias[i & 15], 0.f); }

// ---------------------------------------------------------------------------
extern "C" void kernel_launch(void* const* d_in, const int* in_sizes, int n_in,
                              void* d_out, int out_size, void* d_ws, size_t ws_size,
                              hipStream_t stream)
{
  (void)in_sizes; (void)n_in; (void)out_size; (void)ws_size;
  const int B = 4;
  static const int NODES[5] = {32768, 16384, 8192, 4096, 2048};

  // ---- unpack inputs in setup_inputs() flatten order ---------------------
  int ii = 0;
  auto nf = [&]() { return (const float*)d_in[ii++]; };
  auto ni = [&]() { return (const int*)d_in[ii++]; };
  struct ConvP { const float *w, *b; };
  struct BnP   { const float *s, *h; };
  struct RBP   { ConvP c1; BnP n1; ConvP c2; BnP n2; ConvP cs; BnP ns; bool has; };
  auto rdC = [&]() { ConvP p; p.w = nf(); p.b = nf(); return p; };
  auto rdB = [&]() { BnP p; p.s = nf(); p.h = nf(); return p; };
  auto rdRB = [&](bool has) {
    RBP r; r.c1 = rdC(); r.n1 = rdB(); r.c2 = rdC(); r.n2 = rdB(); r.has = has;
    if (has) { r.cs = rdC(); r.ns = rdB(); }
    else { r.cs.w = nullptr; r.cs.b = nullptr; r.ns.s = nullptr; r.ns.h = nullptr; }
    return r;
  };
  const float* Xin = nf();
  RBP e1 = rdRB(true);  RBP e2 = rdRB(true);  RBP e3 = rdRB(true);
  RBP e4 = rdRB(true);  RBP e5 = rdRB(false);
  ConvP fcmu = rdC();
  ConvP decfc = rdC();
  ConvP up4 = rdC(); RBP d4 = rdRB(false);
  ConvP up3 = rdC(); RBP d3 = rdRB(false);
  ConvP up2c = rdC(); RBP d2 = rdRB(false);
  ConvP up1 = rdC(); RBP d1 = rdRB(false);
  ConvP clast = rdC();
  ConvP glin = rdC();
  const float* chW[5]; const float* chBias[4];
  for (int k = 0; k < 4; ++k) { chW[k] = nf(); chBias[k] = nf(); }
  chW[4] = nf();                                // last cheb bias=None (skipped)
  const float* An[4]; for (int l = 0; l < 4; ++l) An[l] = nf();
  const float* AnL = nf();
  const float* Pv[4]; for (int j = 0; j < 4; ++j) Pv[j] = nf();
  const int *Ar[4], *Ac[4];
  for (int l = 0; l < 4; ++l) { Ar[l] = ni(); Ac[l] = ni(); }
  const int* ALr = ni(); const int* ALc = ni();
  const int *Pr[4], *Pc[4];
  for (int j = 0; j < 4; ++j) { Pr[j] = ni(); Pc[j] = ni(); }

  // ---- workspace layout ---------------------------------------------------
  size_t off = 0;
  auto alloc = [&](size_t bytes) {
    void* p = (char*)d_ws + off; off += (bytes + 255) & ~(size_t)255; return p;
  };
  half_t*   xin   = (half_t*)alloc((size_t)B * 512 * 512 * 8  * 2);
  half_t*   c1    = (half_t*)alloc((size_t)B * 512 * 512 * 8  * 2);
  half_t*   c2    = (half_t*)alloc((size_t)B * 256 * 256 * 16 * 2);
  half_t*   c3    = (half_t*)alloc((size_t)B * 128 * 128 * 32 * 2);
  half_t*   c4    = (half_t*)alloc((size_t)B * 64  * 64  * 64 * 2);
  half_t*   h5    = (half_t*)alloc((size_t)B * 32  * 32  * 64 * 2);
  half_t*   bufA  = (half_t*)alloc((size_t)B * 512 * 512 * 16 * 2);
  half_t*   bufB  = (half_t*)alloc((size_t)B * 512 * 512 * 16 * 2);
  half_t*   bufC  = (half_t*)alloc((size_t)B * 512 * 512 * 16 * 2);
  uint32_t* wpack = (uint32_t*)alloc((size_t)4 * 9 * 2 * 256 * 4);  // 4 chunks max
  float*    mu    = (float*)alloc(256 * 4);
  float*    gb[6];
  for (int k = 0; k < 6; ++k) gb[k] = (float*)alloc((size_t)B * 32768 * 16 * 4);

  // ---- launch helpers -----------------------------------------------------
  auto lg2 = [](int v) { int s = 0; while ((1 << s) < v) ++s; return s; };

  auto conv = [&](const half_t* in, ConvP cp, const float* sc, const float* sh,
                  const half_t* skip, half_t* out, int H, int W,
                  int CinBuf, int CinW, int Cout, int ksz, int relu) {
    const int tilesW = W >> 4;
    const int tshift = lg2(tilesW);
    const int tilesPerImg = H * tilesW;
    const int coChunks = (Cout + 15) / 16;
    dim3 grid((tilesPerImg + 7) / 8, coChunks, B);
#define CONV_CASE(KS, CB)                                                     \
    do {                                                                      \
      pack_weights_kernel<KS, CB><<<coChunks, 256, 0, stream>>>(              \
          cp.w, wpack, CinW, Cout);                                           \
      conv_wmma_kernel<KS, CB><<<grid, 256, 0, stream>>>(                     \
          in, wpack, cp.b, sc, sh, skip, out, H, W, tshift, Cout, relu);      \
    } while (0)
    if (ksz == 3) {
      if      (CinBuf == 8)  CONV_CASE(3, 8);
      else if (CinBuf == 16) CONV_CASE(3, 16);
      else if (CinBuf == 32) CONV_CASE(3, 32);
      else                   CONV_CASE(3, 64);
    } else {
      if      (CinBuf == 8)  CONV_CASE(1, 8);
      else if (CinBuf == 16) CONV_CASE(1, 16);
      else                   CONV_CASE(1, 32);
    }
#undef CONV_CASE
  };
  auto runRB = [&](const half_t* in, RBP& r, half_t* tH, half_t* tS, half_t* out,
                   int H, int W, int CinBuf, int CinW, int Cout) {
    const half_t* skip = in;
    if (r.has) {
      conv(in, r.cs, r.ns.s, r.ns.h, nullptr, tS, H, W, CinBuf, CinW, Cout, 1, 0);
      skip = tS;
    }
    conv(in, r.c1, r.n1.s, r.n1.h, nullptr, tH, H, W, CinBuf, CinW, Cout, 3, 1);
    conv(tH, r.c2, r.n2.s, r.n2.h, skip, out, H, W, Cout, Cout, Cout, 3, 1);
  };
  auto pool = [&](const half_t* in, half_t* out, int H, int W, int C) {
    int total = B * (H / 2) * (W / 2) * C;
    maxpool_kernel<<<(total + 255) / 256, 256, 0, stream>>>(
        in, out, H, W, C, lg2(C), lg2(W / 2), lg2(H / 2), total);
  };
  auto upsmp = [&](const half_t* in, half_t* out, int H, int W, int C) {
    int total = B * H * 2 * W * 2 * C;
    upsample_kernel<<<(total + 255) / 256, 256, 0, stream>>>(
        in, out, H, W, C, lg2(C), lg2(W * 2), lg2(H * 2), total);
  };
  auto fillF = [&](float* p, float v, int n) {
    fill_kernel<<<(n + 255) / 256, 256, 0, stream>>>(p, v, n);
  };
  auto spmmL = [&](const int* r, const int* cc, const float* v, const float* x, float* o,
                   int E, int F, int nIn, int nOut) {
    dim3 grid((E + 255) / 256, B);
    spmm_kernel<<<grid, 256, 0, stream>>>(r, cc, v, x, o, E, F, nIn, nOut);
  };
  auto chebmmL = [&](const float* T, const float* Wk, float* o, int rows, int G) {
    const int gShift = (G == 16) ? 4 : 1;
    int total = rows << gShift;
    chebmm_kernel<<<(total + 255) / 256, 256, 0, stream>>>(T, Wk, o, rows, G, gShift);
  };
  auto runCheb = [&](const float* gIn, int n, int E,
                     const int* Arr, const int* Acc, const float* Ann,
                     const float* Wc, int G, const float* bias, float* outBuf,
                     float* T1, float* S, float* T2a, float* T2b) {
    int rows = B * n;
    fillF(outBuf, 0.f, rows * G);
    chebmmL(gIn, Wc, outBuf, rows, G);                 // k=0 : T0 = x
    fillF(T1, 0.f, rows * 16);
    spmmL(Arr, Acc, Ann, gIn, T1, E, 16, n, n);        // T1 = S x
    chebmmL(T1, Wc + 16 * G, outBuf, rows, G);         // k=1
    const float* Tp = gIn;
    const float* Tc = T1;
    float* slots[3] = { T2a, T2b, T1 };
    int s = 0;
    for (int k = 2; k < 6; ++k) {
      float* Tn = slots[s]; s = (s + 1) % 3;
      fillF(S, 0.f, rows * 16);
      spmmL(Arr, Acc, Ann, Tc, S, E, 16, n, n);
      { int total = rows * 16;
        cheb_combine_kernel<<<(total + 255) / 256, 256, 0, stream>>>(S, Tp, Tn, total); }
      chebmmL(Tn, Wc + k * 16 * G, outBuf, rows, G);
      Tp = Tc; Tc = Tn;
    }
    if (bias) {
      int total = rows * G;
      bias_relu_kernel<<<(total + 255) / 256, 256, 0, stream>>>(outBuf, bias, total);
    }
  };

  // ======================= encoder ========================================
  { int total = B * 512 * 512;
    pack_input_kernel<<<(total + 255) / 256, 256, 0, stream>>>(Xin, xin, total); }
  runRB(xin, e1, bufA, bufB, c1, 512, 512, 8, 1, 8);
  pool(c1, bufC, 512, 512, 8);
  runRB(bufC, e2, bufA, bufB, c2, 256, 256, 8, 8, 16);
  pool(c2, bufC, 256, 256, 16);
  runRB(bufC, e3, bufA, bufB, c3, 128, 128, 16, 16, 32);
  pool(c3, bufC, 128, 128, 32);
  runRB(bufC, e4, bufA, bufB, c4, 64, 64, 32, 32, 64);
  pool(c4, bufC, 64, 64, 64);
  runRB(bufC, e5, bufA, bufB, h5, 32, 32, 64, 64, 64);
  fcmu_kernel<<<B * 64, 256, 0, stream>>>(h5, fcmu.w, fcmu.b, mu);

  // ======================= GNN decoder ====================================
  { int total = B * 32768;
    declin_kernel<<<(total + 255) / 256, 256, 0, stream>>>(mu, glin.w, glin.b, gb[0]); }
  float* gCur = gb[0];
  for (int i = 0; i < 4; ++i) {
    int lev = 3 - i;
    int nOut = NODES[lev], nIn = NODES[lev + 1];
    float* gP = gb[1];
    fillF(gP, 0.f, B * nOut * 16);
    spmmL(Pr[lev], Pc[lev], Pv[lev], gCur, gP, 3 * nOut, 16, nIn, nOut);
    float* outb = gb[2];
    runCheb(gP, nOut, 8 * nOut, Ar[lev], Ac[lev], An[lev], chW[i], 16, chBias[i],
            outb, gb[3], gb[4], gb[5], gb[0]);
    gCur = outb;
  }
  runCheb(gCur, 32768, 8 * 32768, ALr, ALc, AnL, chW[4], 2, nullptr,
          (float*)d_out, gb[3], gb[4], gb[5], gb[0]);

  // ======================= CNN decoder ====================================
  { int total = B * 65536;
    decfc_kernel<<<(total + 255) / 256, 256, 0, stream>>>(mu, decfc.w, decfc.b, bufA); }
  upsmp(bufA, bufB, 32, 32, 64);
  conv(bufB, up4, nullptr, nullptr, c4, bufA, 64, 64, 64, 64, 64, 3, 0);
  runRB(bufA, d4, bufB, nullptr, bufC, 64, 64, 64, 64, 64);
  upsmp(bufC, bufA, 64, 64, 64);
  conv(bufA, up3, nullptr, nullptr, c3, bufB, 128, 128, 64, 64, 32, 3, 0);
  runRB(bufB, d3, bufA, nullptr, bufC, 128, 128, 32, 32, 32);
  upsmp(bufC, bufA, 128, 128, 32);
  conv(bufA, up2c, nullptr, nullptr, c2, bufB, 256, 256, 32, 32, 16, 3, 0);
  runRB(bufB, d2, bufA, nullptr, bufC, 256, 256, 16, 16, 16);
  upsmp(bufC, bufA, 256, 256, 16);
  conv(bufA, up1, nullptr, nullptr, c1, bufB, 512, 512, 16, 16, 8, 3, 0);
  runRB(bufB, d1, bufA, nullptr, bufC, 512, 512, 8, 8, 8);
  { int total = 4 * 4 * 512 * 512;
    conv_last_kernel<<<(total + 255) / 256, 256, 0, stream>>>(
        bufC, clast.w, clast.b, (float*)d_out + 262144); }
}
